// LuongAttention_23467701305764
// MI455X (gfx1250) — matrix operand order repaired and checked by reading, not compiled, not run on hardware
//
#include <hip/hip_runtime.h>

#define BATCH 32
#define TQ    1024
#define TVDIM 1024
#define DIM   512
#define SROW  (TVDIM + 4)   // padded fp32 score row stride (floats)
#define PROW  (TVDIM + 8)   // padded bf16 prob row stride (elements, 16B mult)

typedef __attribute__((ext_vector_type(16))) __bf16        v16bf;
typedef __attribute__((ext_vector_type(8)))  float         v8f;
typedef __attribute__((ext_vector_type(8)))  unsigned int  v8u;

// ---------- bf16 packing ----------
// RNE pack (used only in one-time staging passes).
static __device__ __forceinline__ unsigned int pk2(float a, float b) {
  unsigned int x = __builtin_bit_cast(unsigned int, a);
  unsigned int y = __builtin_bit_cast(unsigned int, b);
  x = (x + 0x7FFFu + ((x >> 16) & 1u)) >> 16;
  y = (y + 0x7FFFu + ((y >> 16) & 1u)) >> 16;
  return (y << 16) | (x & 0xFFFFu);
}
static __device__ __forceinline__ unsigned short f2bf(float a) {
  unsigned int x = __builtin_bit_cast(unsigned int, a);
  x = (x + 0x7FFFu + ((x >> 16) & 1u)) >> 16;
  return (unsigned short)x;
}
// Truncating pack: one v_perm_b32 — result = {hi16(b), hi16(a)}.
static __device__ __forceinline__ unsigned int pkt2(float a, float b) {
  return __builtin_amdgcn_perm(__builtin_bit_cast(unsigned int, b),
                               __builtin_bit_cast(unsigned int, a), 0x07060302u);
}

static __device__ __forceinline__ v16bf as_bf(v8u p) {
  return __builtin_bit_cast(v16bf, p);
}

// A operand (16x32 bf16): 16 contiguous K[koff..+15] at p (koff = 16*half
// folded into p). Two 16B loads (global or LDS).
static __device__ __forceinline__ v16bf load_a_bf16(const unsigned short* p) {
  const uint4 lo = *reinterpret_cast<const uint4*>(p);
  const uint4 hi = *reinterpret_cast<const uint4*>(p + 8);
  v8u r;
  r[0] = lo.x; r[1] = lo.y; r[2] = lo.z; r[3] = lo.w;
  r[4] = hi.x; r[5] = hi.y; r[6] = hi.z; r[7] = hi.w;
  return as_bf(r);
}
// B operand (32x16): 16 contiguous bf16 (32 bytes, 32B-aligned).
static __device__ __forceinline__ v16bf load_b_bf16(const unsigned short* p) {
  return as_bf(*reinterpret_cast<const v8u*>(p));
}
static __device__ __forceinline__ v8f wmma_bf16(v16bf a, v16bf b, v8f c) {
  return __builtin_amdgcn_wmma_f32_16x16x32_bf16(false, a, false, b, (short)0, c,
                                                 false, false);
}

// ---------------------------------------------------------------------------
// Kernel 0: streaming fp32 -> bf16 convert (8 elements / thread, RNE)
// ---------------------------------------------------------------------------
extern "C" __global__ void __launch_bounds__(256)
k_f32_to_bf16(const float* __restrict__ in, unsigned short* __restrict__ outp,
              long long n8) {
  const long long i = (long long)blockIdx.x * 256 + threadIdx.x;
  if (i >= n8) return;
  const float4* p = reinterpret_cast<const float4*>(in) + 2 * i;
  const float4 a = p[0], b = p[1];
  uint4 o;
  o.x = pk2(a.x, a.y); o.y = pk2(a.z, a.w);
  o.z = pk2(b.x, b.y); o.w = pk2(b.z, b.w);
  reinterpret_cast<uint4*>(outp)[i] = o;
}

// ---------------------------------------------------------------------------
// Kernel 1: V [B,TV,D] fp32  ->  V^T [B,D,TV] bf16   (tiled LDS transpose)
// ---------------------------------------------------------------------------
extern "C" __global__ void __launch_bounds__(256)
k_transpose_v(const float* __restrict__ val, unsigned short* __restrict__ vT) {
  __shared__ float tile[32][33];
  const int bi = blockIdx.z;
  const int v0 = blockIdx.x << 5;
  const int d0 = blockIdx.y << 5;
  const int tx = threadIdx.x & 31;
  const int ty = threadIdx.x >> 5;
#pragma unroll
  for (int i = ty; i < 32; i += 8)
    tile[i][tx] = val[((size_t)bi * TVDIM + v0 + i) * DIM + d0 + tx];
  __syncthreads();
#pragma unroll
  for (int i = ty; i < 32; i += 8)
    vT[((size_t)bi * DIM + d0 + i) * TVDIM + v0 + tx] = f2bf(tile[tx][i]);
}

// ---------------------------------------------------------------------------
// Kernel 2: keys = V @ W^T + bias  ->  bf16 [B,TV,D]
// One wave: 16(v) x 64(e) slab, A chunk reused over 4 B tiles.
// A: pre-converted value bf16. B: pre-converted W bf16. No in-loop cvt.
// ---------------------------------------------------------------------------
extern "C" __global__ void __launch_bounds__(256)
k_keys_gemm(const unsigned short* __restrict__ vbf,
            const unsigned short* __restrict__ wbf,
            const float* __restrict__ bias, unsigned short* __restrict__ keysbf) {
  const int lane = threadIdx.x & 31;
  const int wid  = (blockIdx.x << 3) + (threadIdx.x >> 5);  // 16384 wave tasks
  const int et = wid & 7;          // e-tile of 64
  const int vt = (wid >> 3) & 63;  // v-tile of 16
  const int bi = wid >> 9;
  const int e0 = et << 6, v0 = vt << 4;
  const int col = lane & 15, hi = lane >> 4, mb = hi << 3;

  v8f acc[4] = {};
  const unsigned short* arow =
      vbf + ((size_t)bi * TVDIM + v0 + col) * DIM + hi * 16;
  for (int kc = 0; kc < DIM; kc += 32) {
    v16bf A = load_a_bf16(arow + kc);
#pragma unroll
    for (int nt = 0; nt < 4; ++nt) {
      const unsigned short* brow =
          wbf + (size_t)(e0 + nt * 16 + col) * DIM + hi * 16 + kc;
      acc[nt] = wmma_bf16(A, load_b_bf16(brow), acc[nt]);
    }
  }
#pragma unroll
  for (int nt = 0; nt < 4; ++nt) {
    const int e = e0 + nt * 16 + col;
    const float bv = bias[e];
#pragma unroll
    for (int r = 0; r < 8; ++r)
      keysbf[((size_t)bi * TVDIM + v0 + mb + r) * DIM + e] = f2bf(acc[nt][r] + bv);
  }
}

// ---------------------------------------------------------------------------
// Kernel 3 (fused): per 16-query tile: scores (WMMA) -> softmax in LDS
// (fp32 S + bf16 P written in the same pass) -> alignment^T -> context.
// Block = 128 threads (4 waves). LDS: S 16xSROW fp32 + Pbf 16xPROW bf16 ~97KB.
// ---------------------------------------------------------------------------
extern "C" __global__ void __launch_bounds__(128)
k_attn_fused(const unsigned short* __restrict__ qbf,
             const unsigned short* __restrict__ keysbf,
             const unsigned short* __restrict__ vT, float* __restrict__ out) {
  extern __shared__ float S[];                         // [16][SROW] fp32
  unsigned short* Pbf = (unsigned short*)(S + 16 * SROW);  // [16][PROW] bf16
  const int bi = blockIdx.x >> 6;
  const int q0 = (blockIdx.x & 63) << 4;
  const int lane = threadIdx.x & 31;
  const int wave = threadIdx.x >> 5;
  const int col = lane & 15, hi = lane >> 4, mb = hi << 3;

  // ---- Phase 1: S[16][TV] = Q_tile @ keys^T ----
  // Wave owns kv tiles j = wave + 4t, grouped 4-wide (A chunk feeds 4 WMMAs).
  const unsigned short* qrow =
      qbf + ((size_t)bi * TQ + q0 + col) * DIM + hi * 16;
  for (int g = 0; g < 4; ++g) {
    v8f acc[4] = {};
    for (int kc = 0; kc < DIM; kc += 32) {
      v16bf A = load_a_bf16(qrow + kc);
#pragma unroll
      for (int u = 0; u < 4; ++u) {
        const int j = wave + 16 * g + 4 * u;  // kv tile 0..63
        const unsigned short* krow =
            keysbf + ((size_t)bi * TVDIM + j * 16 + col) * DIM + hi * 16 + kc;
        acc[u] = wmma_bf16(A, load_b_bf16(krow), acc[u]);
      }
    }
#pragma unroll
    for (int u = 0; u < 4; ++u) {
      const int n = (wave + 16 * g + 4 * u) * 16 + col;
#pragma unroll
      for (int r = 0; r < 8; ++r) S[(size_t)(mb + r) * SROW + n] = acc[u][r];
    }
  }
  __syncthreads();

  // ---- Phase 2: row softmax over TV; also emit bf16 P (v_perm pack) ----
  {
    const int row = threadIdx.x >> 3, sub = threadIdx.x & 7;
    float* sr = S + (size_t)row * SROW + sub * 128;
    float m = -3.402823466e38f;
    for (int i = 0; i < 128; ++i) m = fmaxf(m, sr[i]);
#pragma unroll
    for (int o = 1; o < 8; o <<= 1) m = fmaxf(m, __shfl_xor(m, o, 32));
    float s = 0.f;
    for (int i = 0; i < 128; ++i) { float e = __expf(sr[i] - m); sr[i] = e; s += e; }
#pragma unroll
    for (int o = 1; o < 8; o <<= 1) s += __shfl_xor(s, o, 32);
    const float inv = 1.f / s;
    unsigned int* pr =
        reinterpret_cast<unsigned int*>(Pbf + (size_t)row * PROW) + sub * 64;
    for (int i = 0; i < 64; ++i) {
      const float e0 = sr[2 * i] * inv;
      const float e1 = sr[2 * i + 1] * inv;
      sr[2 * i] = e0;
      sr[2 * i + 1] = e1;
      pr[i] = pkt2(e0, e1);
    }
  }
  __syncthreads();

  // ---- Phase 2b: alignment^T[b][v][q0+qq] = P[qq][v] (fp32 from S) ----
  {
    float* alb = out + (size_t)BATCH * TQ * 2 * DIM + (size_t)bi * TVDIM * TQ + q0;
    const int qq = threadIdx.x & 15;
    for (int v = threadIdx.x >> 4; v < TVDIM; v += 8)
      alb[(size_t)v * TQ + qq] = S[(size_t)qq * SROW + v];
  }

  // ---- Phase 3: context = P @ V; each wave owns 128 feature cols ----
  {
    const int d0 = wave << 7;
    v8f acc[8] = {};
    const unsigned short* arow = Pbf + (size_t)col * PROW + hi * 16;  // P row
    for (int kc = 0; kc < TVDIM; kc += 32) {
      v16bf A = load_a_bf16(arow + kc);  // pure ds_load_b128 pair
#pragma unroll
      for (int nt = 0; nt < 8; ++nt) {
        const unsigned short* bp =
            vT + ((size_t)bi * DIM + d0 + nt * 16 + col) * TVDIM + hi * 16 + kc;
        acc[nt] = wmma_bf16(A, load_b_bf16(bp), acc[nt]);
      }
    }
    float* cb = out + (size_t)bi * TQ * 2 * DIM;
#pragma unroll
    for (int nt = 0; nt < 8; ++nt)
#pragma unroll
      for (int r = 0; r < 8; ++r)
        cb[(size_t)(q0 + mb + r) * (2 * DIM) + d0 + nt * 16 + col] = acc[nt][r];
  }
}

// ---------------------------------------------------------------------------
// Kernel 4: out[b][q][D + d] = query[b][q][d]   (concat second half)
// ---------------------------------------------------------------------------
extern "C" __global__ void __launch_bounds__(256)
k_copy_q(const float* __restrict__ q, float* __restrict__ out) {
  const size_t i = (size_t)blockIdx.x * 256 + threadIdx.x;
  if (i >= (size_t)BATCH * TQ * DIM) return;
  const size_t d = i % DIM, r = i / DIM;
  out[r * (2 * DIM) + DIM + d] = q[i];
}

// ---------------------------------------------------------------------------
extern "C" void kernel_launch(void* const* d_in, const int* in_sizes, int n_in,
                              void* d_out, int out_size, void* d_ws, size_t ws_size,
                              hipStream_t stream) {
  const float* query  = (const float*)d_in[0];  // [B,TQ,D]
  const float* value  = (const float*)d_in[1];  // [B,TV,D]
  const float* weight = (const float*)d_in[2];  // [D,D]
  const float* bias   = (const float*)d_in[3];  // [D]
  float* out = (float*)d_out;

  // Workspace layout (bf16 staging):
  unsigned short* vT     = (unsigned short*)d_ws;                 // [B,D,TV]
  unsigned short* keysbf = vT + (size_t)BATCH * DIM * TVDIM;      // [B,TV,D]
  unsigned short* qbf    = keysbf + (size_t)BATCH * TVDIM * DIM;  // [B,TQ,D]
  unsigned short* vbf    = qbf + (size_t)BATCH * TQ * DIM;        // [B,TV,D]
  unsigned short* wbf    = vbf + (size_t)BATCH * TVDIM * DIM;     // [D,D]

  const int smem =
      16 * SROW * (int)sizeof(float) + 16 * PROW * (int)sizeof(unsigned short);
  hipFuncSetAttribute(reinterpret_cast<const void*>(&k_attn_fused),
                      hipFuncAttributeMaxDynamicSharedMemorySize, smem);

  const long long nq8 = (long long)BATCH * TQ * DIM / 8;
  const long long nv8 = (long long)BATCH * TVDIM * DIM / 8;
  const long long nw8 = (long long)DIM * DIM / 8;
  k_f32_to_bf16<<<(int)(nq8 / 256), 256, 0, stream>>>(query, qbf, nq8);
  k_f32_to_bf16<<<(int)(nv8 / 256), 256, 0, stream>>>(value, vbf, nv8);
  k_f32_to_bf16<<<(int)(nw8 / 256), 256, 0, stream>>>(weight, wbf, nw8);

  dim3 gt(TVDIM / 32, DIM / 32, BATCH);
  k_transpose_v<<<gt, 256, 0, stream>>>(value, vT);
  k_keys_gemm<<<(BATCH * (TVDIM / 16) * (DIM / 64)) / 8, 256, 0, stream>>>(
      vbf, wbf, bias, keysbf);
  k_attn_fused<<<BATCH * (TQ / 16), 128, smem, stream>>>(qbf, keysbf, vT, out);
  k_copy_q<<<(BATCH * TQ * DIM) / 256, 256, 0, stream>>>(query, out);
}
